// DenseSNN_24249385353885
// MI455X (gfx1250) — matrix operand order repaired and checked by reading, not compiled, not run on hardware
//
#include <hip/hip_runtime.h>

// ---------------------------------------------------------------------------
// DenseSNN on MI455X (gfx1250): WMMA f16 with hi/lo split for fp32 accuracy.
//  - Layer-1 currents hoisted into one big parallel GEMM (workspace allowing),
//    with x pre-split into f16 hi/lo so the GEMM inner loop is pure WMMA.
//  - Sequential part: 3 spike GEMMs/step (A is exact {0,1} in f16), LIF fused
//    into the GEMM epilogue.
// ---------------------------------------------------------------------------

typedef _Float16 v16h __attribute__((ext_vector_type(16)));
typedef float    v8f  __attribute__((ext_vector_type(8)));
using hf = _Float16;

static constexpr int Tn   = 100;
static constexpr int Bsz  = 128;
static constexpr int Dn   = 2048;
static constexpr int Hn   = 2048;
static constexpr int Cout = 100;
static constexpr int Cpad = 128;

// A-operand modes for the GEMM kernel
static constexpr int A_F16       = 0;  // single f16 matrix (spikes; exact)
static constexpr int A_F16_SPLIT = 1;  // pre-split hi/lo f16 matrices
static constexpr int A_F32_SPLIT = 2;  // fp32, split on the fly (fallback)

// -------------------------------- device helpers ---------------------------

__device__ __forceinline__ v8f wmma_f16(v16h a, v16h b, v8f c) {
  // D = A(16x32 f16) * B(32x16 f16) + C(16x16 f32)
  return __builtin_amdgcn_wmma_f32_16x16x32_f16(
      /*neg_a=*/false, a, /*neg_b=*/false, b,
      /*c_mod=*/(short)0, c, /*reuse_a=*/false, /*reuse_b=*/false);
}

// B operand pre-swizzled:
// halves[((kb*NB + nb)*32 + lane)*16 + h] = W[kb*32 + (lane>=16)*8 + (h&7) + (h>>3)*16][nb*16 + (lane&15)]
__device__ __forceinline__ v16h load_b_frag(const hf* __restrict__ Bws,
                                            int kb, int nb, int NB, int lane) {
  const hf* p = Bws + (((size_t)kb * NB + nb) * 32 + lane) * 16;
  union { v16h v; uint4 u[2]; } f;
  const uint4* q = (const uint4*)p;
  f.u[0] = q[0];
  f.u[1] = q[1];
  return f.v;
}

// A operand, f16 row-major [*, K]: lane holds row = lane&15,
// halves 0..7 -> K = kb*32 + kbase + 0..7 ; halves 8..15 -> +16..+23
// (kbase = (lane>=16)*8)  — matches ISA 16-bit A-matrix 16x32 layout.
__device__ __forceinline__ v16h load_a_f16(const hf* __restrict__ A, int K,
                                           int kb, int lane) {
  const int r = lane & 15;
  const int kbase = (lane >> 4) * 8;
  const hf* p = A + (size_t)r * K + kb * 32 + kbase;
  union { v16h v; uint4 u[2]; } f;
  f.u[0] = *(const uint4*)(p);
  f.u[1] = *(const uint4*)(p + 16);
  return f.v;
}

// A operand, fp32 row-major, split into hi/lo f16 on the fly (fallback only).
__device__ __forceinline__ void load_a_f32_split(const float* __restrict__ A,
                                                 int K, int kb, int lane,
                                                 v16h& hi, v16h& lo) {
  const int r = lane & 15;
  const int kbase = (lane >> 4) * 8;
  const float* p = A + (size_t)r * K + kb * 32 + kbase;
  float f[16];
  const float4* q0 = (const float4*)p;
  const float4* q1 = (const float4*)(p + 16);
  float4 a0 = q0[0], a1 = q0[1], b0 = q1[0], b1 = q1[1];
  f[0]=a0.x; f[1]=a0.y; f[2]=a0.z; f[3]=a0.w;
  f[4]=a1.x; f[5]=a1.y; f[6]=a1.z; f[7]=a1.w;
  f[8]=b0.x; f[9]=b0.y; f[10]=b0.z; f[11]=b0.w;
  f[12]=b1.x; f[13]=b1.y; f[14]=b1.z; f[15]=b1.w;
  union { v16h v; hf e[16]; } uh, ul;
#pragma unroll
  for (int j = 0; j < 16; ++j) {
    hf h = (hf)f[j];
    uh.e[j] = h;
    ul.e[j] = (hf)(f[j] - (float)h);
  }
  hi = uh.v;
  lo = ul.v;
}

// -------------------------------- kernels ----------------------------------

// Pre-swizzle fp32 weights into WMMA-B-friendly hi/lo f16 tiles (zero-pad cols).
__global__ void conv_swizzle_kernel(const float* __restrict__ W, int K, int N,
                                    int Npad, hf* __restrict__ hi,
                                    hf* __restrict__ lo) {
  int i = blockIdx.x * blockDim.x + threadIdx.x;
  int total = K * Npad;
  if (i >= total) return;
  int k = i / Npad, n = i - k * Npad;
  float w = (n < N) ? W[(size_t)k * N + n] : 0.0f;
  hf h = (hf)w;
  hf l = (hf)(w - (float)h);
  int kb = k >> 5, kl = k & 31, nb = n >> 4, nl = n & 15;
  int lane = (((kl >> 3) & 1) << 4) | nl;
  int hh = (kl & 7) | ((kl >> 4) << 3);
  size_t off = (((size_t)kb * (Npad >> 4) + nb) * 32 + lane) * 16 + hh;
  hi[off] = h;
  lo[off] = l;
}

// Split fp32 activations into hi/lo f16 (row-major, elementwise, bandwidth-bound).
__global__ void split_f32_kernel(const float* __restrict__ X,
                                 hf* __restrict__ hi, hf* __restrict__ lo,
                                 size_t n) {
  size_t i = (size_t)blockIdx.x * blockDim.x + threadIdx.x;
  if (i >= n) return;
  float v = X[i];
  hf h = (hf)v;
  hi[i] = h;
  lo[i] = (hf)(v - (float)h);
}

// GEMM (+ optional fused LIF). Block = 128 thr (4 waves); each wave owns two
// 16x16 output tiles -> block covers 16 rows x 128 cols. K multiple of 32.
template <int AMODE, bool OUTLAYER, bool STORECUR>
__global__ void gemm_lif_kernel(const void* __restrict__ Aptr,
                                const void* __restrict__ Aptr2,
                                const hf* __restrict__ Bhi,
                                const hf* __restrict__ Blo,
                                const float* __restrict__ bias, int biasN,
                                int K, int N, float* __restrict__ mem,
                                hf* __restrict__ spkOut,
                                float* __restrict__ accOut) {
  const int lane = threadIdx.x & 31;
  const int wave = threadIdx.x >> 5;
  const int mtile = blockIdx.y;
  const int nt0 = blockIdx.x * 8 + wave * 2;
  const int NB = N >> 4;
  const int row0 = mtile * 16;
  const int kblocks = K >> 5;

  const float* A32 = nullptr;
  const hf* A16 = nullptr;
  const hf* A16b = nullptr;
  if constexpr (AMODE == A_F32_SPLIT) {
    A32 = (const float*)Aptr + (size_t)row0 * K;
  } else {
    A16 = (const hf*)Aptr + (size_t)row0 * K;
    if constexpr (AMODE == A_F16_SPLIT)
      A16b = (const hf*)Aptr2 + (size_t)row0 * K;
  }

  v8f acc[2] = {};

  for (int kb = 0; kb < kblocks; ++kb) {
    v16h ahi, alo{};
    if constexpr (AMODE == A_F32_SPLIT) {
      load_a_f32_split(A32, K, kb, lane, ahi, alo);
    } else {
      ahi = load_a_f16(A16, K, kb, lane);
      if constexpr (AMODE == A_F16_SPLIT) alo = load_a_f16(A16b, K, kb, lane);
    }

    v16h bh0 = load_b_frag(Bhi, kb, nt0,     NB, lane);
    v16h bh1 = load_b_frag(Bhi, kb, nt0 + 1, NB, lane);
    v16h bl0 = load_b_frag(Blo, kb, nt0,     NB, lane);
    v16h bl1 = load_b_frag(Blo, kb, nt0 + 1, NB, lane);

    if (kb + 1 < kblocks) {  // pull next B K-tile toward WGP (global_prefetch_b8)
      __builtin_prefetch(Bhi + (((size_t)(kb + 1) * NB + nt0) * 32 + lane) * 16, 0, 1);
      __builtin_prefetch(Blo + (((size_t)(kb + 1) * NB + nt0) * 32 + lane) * 16, 0, 1);
    }

    acc[0] = wmma_f16(ahi, bh0, acc[0]);
    acc[1] = wmma_f16(ahi, bh1, acc[1]);
    acc[0] = wmma_f16(ahi, bl0, acc[0]);
    acc[1] = wmma_f16(ahi, bl1, acc[1]);
    if constexpr (AMODE != A_F16) {
      acc[0] = wmma_f16(alo, bh0, acc[0]);
      acc[1] = wmma_f16(alo, bh1, acc[1]);
    }
  }

  // Epilogue: C/D layout -> element (M = (lane>=16)*8 + r, N = lane&15) in VGPR r.
  const int nl = lane & 15;
  const int mb = (lane >> 4) * 8;
#pragma unroll
  for (int t = 0; t < 2; ++t) {
    const int col = (nt0 + t) * 16 + nl;
    const float bv = (col < biasN) ? bias[col] : 0.0f;
#pragma unroll
    for (int r = 0; r < 8; ++r) {
      const int brow = row0 + mb + r;
      const size_t idx = (size_t)brow * N + col;
      float cur = acc[t][r] + bv;
      if constexpr (STORECUR) {
        accOut[idx] = cur;
      } else {
        float mp = mem[idx];
        float reset = (mp > 1.0f) ? 1.0f : 0.0f;     // thresh = 1
        float mnew = 0.9f * mp + cur - reset;        // beta = 0.9, reset-by-subtract
        mem[idx] = mnew;
        float spk = (mnew > 1.0f) ? 1.0f : 0.0f;
        if (spkOut) spkOut[idx] = (hf)spk;
        if constexpr (OUTLAYER) accOut[idx] += spk;
      }
    }
  }
}

// Elementwise LIF for precomputed layer-1 currents.
__global__ void lif_elem_kernel(const float* __restrict__ cur,
                                float* __restrict__ mem, hf* __restrict__ spk,
                                int n) {
  int i = blockIdx.x * blockDim.x + threadIdx.x;
  if (i >= n) return;
  float c = cur[i];
  float mp = mem[i];
  float reset = (mp > 1.0f) ? 1.0f : 0.0f;
  float mnew = 0.9f * mp + c - reset;
  mem[i] = mnew;
  spk[i] = (hf)((mnew > 1.0f) ? 1.0f : 0.0f);
}

__global__ void copy_out_kernel(const float* __restrict__ sum,
                                float* __restrict__ out) {
  int i = blockIdx.x * blockDim.x + threadIdx.x;
  if (i >= Bsz * Cout) return;
  int b = i / Cout, c = i - b * Cout;
  out[i] = sum[(size_t)b * Cpad + c];
}

// -------------------------------- host launcher ----------------------------

extern "C" void kernel_launch(void* const* d_in, const int* in_sizes, int n_in,
                              void* d_out, int out_size, void* d_ws,
                              size_t ws_size, hipStream_t stream) {
  (void)in_sizes; (void)n_in; (void)out_size;
  const float* x_seq = (const float*)d_in[0];
  const float* W1 = (const float*)d_in[1];
  const float* b1 = (const float*)d_in[2];
  const float* W2 = (const float*)d_in[3];
  const float* b2 = (const float*)d_in[4];
  const float* W3 = (const float*)d_in[5];
  const float* b3 = (const float*)d_in[6];
  const float* Wo = (const float*)d_in[7];
  const float* bo = (const float*)d_in[8];

  char* ws = (char*)d_ws;
  size_t off = 0;
  auto alloc = [&](size_t bytes) -> size_t {
    size_t o = off;
    off += (bytes + 255) & ~(size_t)255;
    return o;
  };

  const size_t wBytes = (size_t)Dn * Hn * sizeof(hf);  // 8 MB per half-matrix
  size_t oW1hi = alloc(wBytes), oW1lo = alloc(wBytes);
  size_t oW2hi = alloc(wBytes), oW2lo = alloc(wBytes);
  size_t oW3hi = alloc(wBytes), oW3lo = alloc(wBytes);
  size_t oWohi = alloc((size_t)Hn * Cpad * sizeof(hf));
  size_t oWolo = alloc((size_t)Hn * Cpad * sizeof(hf));
  size_t oS1 = alloc((size_t)Bsz * Hn * sizeof(hf));
  size_t oS2 = alloc((size_t)Bsz * Hn * sizeof(hf));
  size_t oS3 = alloc((size_t)Bsz * Hn * sizeof(hf));
  size_t oM1 = alloc((size_t)Bsz * Hn * sizeof(float));
  size_t oM2 = alloc((size_t)Bsz * Hn * sizeof(float));
  size_t oM3 = alloc((size_t)Bsz * Hn * sizeof(float));
  size_t oMo = alloc((size_t)Bsz * Cpad * sizeof(float));
  size_t oSum = alloc((size_t)Bsz * Cpad * sizeof(float));

  const size_t xElems = (size_t)Tn * Bsz * Dn;                 // 26.2M
  size_t oCur = alloc(xElems * sizeof(float));                 // 100 MB
  const size_t need_mid = off;
  size_t oXhi = alloc(xElems * sizeof(hf));                    // 50 MB
  size_t oXlo = alloc(xElems * sizeof(hf));                    // 50 MB
  const size_t need_full = off;

  // tier 2: pure-f16 precompute GEMM; tier 1: on-the-fly split precompute;
  // tier 0: no precompute, layer-1 GEMM per step.
  const int tier = (ws_size >= need_full) ? 2 : (ws_size >= need_mid ? 1 : 0);

  hf* W1hi = (hf*)(ws + oW1hi); hf* W1lo = (hf*)(ws + oW1lo);
  hf* W2hi = (hf*)(ws + oW2hi); hf* W2lo = (hf*)(ws + oW2lo);
  hf* W3hi = (hf*)(ws + oW3hi); hf* W3lo = (hf*)(ws + oW3lo);
  hf* Wohi = (hf*)(ws + oWohi); hf* Wolo = (hf*)(ws + oWolo);
  hf* s1 = (hf*)(ws + oS1); hf* s2 = (hf*)(ws + oS2); hf* s3 = (hf*)(ws + oS3);
  float* m1 = (float*)(ws + oM1);
  float* m2 = (float*)(ws + oM2);
  float* m3 = (float*)(ws + oM3);
  float* mo = (float*)(ws + oMo);
  float* sum = (float*)(ws + oSum);
  float* cur1 = (float*)(ws + oCur);
  hf* Xhi = (hf*)(ws + oXhi);
  hf* Xlo = (hf*)(ws + oXlo);

  // Weight conversion + swizzle into WMMA-B layout (hi/lo f16 split).
  {
    int thr = 256;
    conv_swizzle_kernel<<<(Dn * Hn + thr - 1) / thr, thr, 0, stream>>>(
        W1, Dn, Hn, Hn, W1hi, W1lo);
    conv_swizzle_kernel<<<(Hn * Hn + thr - 1) / thr, thr, 0, stream>>>(
        W2, Hn, Hn, Hn, W2hi, W2lo);
    conv_swizzle_kernel<<<(Hn * Hn + thr - 1) / thr, thr, 0, stream>>>(
        W3, Hn, Hn, Hn, W3hi, W3lo);
    conv_swizzle_kernel<<<(Hn * Cpad + thr - 1) / thr, thr, 0, stream>>>(
        Wo, Hn, Cout, Cpad, Wohi, Wolo);
  }

  // Zero membranes + spike sum (contiguous region oM1 .. oSum+size).
  hipMemsetAsync(ws + oM1, 0, (oSum + (size_t)Bsz * Cpad * sizeof(float)) - oM1,
                 stream);

  if (tier == 2) {
    // Split x once; precompute GEMM is then pure f16 WMMA (no VALU in loop).
    split_f32_kernel<<<(unsigned)((xElems + 255) / 256), 256, 0, stream>>>(
        x_seq, Xhi, Xlo, xElems);
    gemm_lif_kernel<A_F16_SPLIT, false, true>
        <<<dim3(Hn / 128, (Tn * Bsz) / 16), 128, 0, stream>>>(
            Xhi, Xlo, W1hi, W1lo, b1, Hn, Dn, Hn, nullptr, nullptr, cur1);
  } else if (tier == 1) {
    gemm_lif_kernel<A_F32_SPLIT, false, true>
        <<<dim3(Hn / 128, (Tn * Bsz) / 16), 128, 0, stream>>>(
            x_seq, nullptr, W1hi, W1lo, b1, Hn, Dn, Hn, nullptr, nullptr, cur1);
  }

  for (int t = 0; t < Tn; ++t) {
    if (tier >= 1) {
      lif_elem_kernel<<<(Bsz * Hn + 255) / 256, 256, 0, stream>>>(
          cur1 + (size_t)t * Bsz * Hn, m1, s1, Bsz * Hn);
    } else {
      gemm_lif_kernel<A_F32_SPLIT, false, false>
          <<<dim3(Hn / 128, Bsz / 16), 128, 0, stream>>>(
              x_seq + (size_t)t * Bsz * Dn, nullptr, W1hi, W1lo, b1, Hn, Dn, Hn,
              m1, s1, nullptr);
    }
    gemm_lif_kernel<A_F16, false, false>
        <<<dim3(Hn / 128, Bsz / 16), 128, 0, stream>>>(
            s1, nullptr, W2hi, W2lo, b2, Hn, Hn, Hn, m2, s2, nullptr);
    gemm_lif_kernel<A_F16, false, false>
        <<<dim3(Hn / 128, Bsz / 16), 128, 0, stream>>>(
            s2, nullptr, W3hi, W3lo, b3, Hn, Hn, Hn, m3, s3, nullptr);
    gemm_lif_kernel<A_F16, true, false>
        <<<dim3(Cpad / 128, Bsz / 16), 128, 0, stream>>>(
            s3, nullptr, Wohi, Wolo, bo, Cout, Hn, Cpad, mo, nullptr, sum);
  }

  copy_out_kernel<<<(Bsz * Cout + 255) / 256, 256, 0, stream>>>(sum,
                                                                (float*)d_out);
}
